// MMDecoderLSTM_19774029431326
// MI455X (gfx1250) — matrix operand order repaired and checked by reading, not compiled, not run on hardware
//
#include <hip/hip_runtime.h>
#include <hip/hip_bf16.h>

// ---------------------------------------------------------------------------
// MMDecoderLSTM for gfx1250: 30-step LSTM decoder, B=65536, H=128, 4H=512.
// Roofline: ~300 GFLOP bf16-WMMA vs ~165MB HBM (7us) -- on-chip bound, and
// the binding on-chip resource is LDS bandwidth for WMMA B-fragments.
// Design:
//   * M=32 rows per wave: one B fragment feeds TWO WMMAs -> LDS weight
//     traffic per row halved vs M=16
//   * 4-wave (128-thread) workgroups: 1 wave/SIMD, full VGPR file per wave,
//     eliminating the round-2 scratch spills (c-state alone is 128 VGPRs).
//     CDNA5 tracks 16-bit WMMA as TRANS, so VALU co-executes with the matrix
//     pipe within a single wave -- no second wave needed for overlap.
//   * weights bf16, pre-swizzled ONCE into LDS in exact B-fragment lane order
//   * h / dec_in round-trip through wave-private LDS (C->A layout transmute;
//     DS ops are in-order within a wave, no barriers needed)
//   * gate nonlinearities via v_tanh_f32 (gfx1250 TRANS op), sequential
//     consumption (i,g -> f -> o) to cap live accumulators
//   * pred_traj scatter-stored; 94MB output < 192MB L2 so stores coalesce
//     into full lines before HBM writeback
// ---------------------------------------------------------------------------

typedef __attribute__((ext_vector_type(16))) __bf16 v16bf;
typedef __attribute__((ext_vector_type(8)))  __bf16 v8bf;
typedef __attribute__((ext_vector_type(8)))  float  v8f;

#define DEV __device__ __forceinline__

DEV float tanh_t(float x) {
#if __has_builtin(__builtin_amdgcn_tanhf)
    return __builtin_amdgcn_tanhf(x);
#elif __has_builtin(__builtin_amdgcn_tanh_f32)
    return __builtin_amdgcn_tanh_f32(x);
#else
    float r;
    // TRANS op: keep 1 op of spacing after result per ISA co-exec hazard rule
    asm("v_tanh_f32 %0, %1\n\tv_nop" : "=v"(r) : "v"(x));
    return r;
#endif
}
DEV float sigmoid_t(float x) { return 0.5f * tanh_t(0.5f * x) + 0.5f; }
DEV float leaky_(float x)    { return x > 0.0f ? x : 0.01f * x; }

DEV v16bf pack16(v8bf lo, v8bf hi) {
    v16bf r;
#pragma unroll
    for (int i = 0; i < 8; ++i) { r[i] = lo[i]; r[i + 8] = hi[i]; }
    return r;
}
// B fragment: lane's 16 bf16 contiguous in LDS (pre-swizzled) -> 2x ds_load_b128
DEV v16bf ldB(const __bf16* p) {
    return pack16(*(const v8bf*)p, *(const v8bf*)(p + 8));
}
// A fragment: two contiguous 8-element K runs (ISA 7.12.2, 16-bit A 16x32)
DEV v16bf ldA(const __bf16* p0, const __bf16* p1) {
    return pack16(*(const v8bf*)p0, *(const v8bf*)p1);
}
DEV v8f wmma_bf16(v16bf a, v16bf b, v8f c) {
    return __builtin_amdgcn_wmma_f32_16x16x32_bf16(
        false, a, false, b, (short)0, c, false, false);
}

// ------------------------- LDS layout (dynamic) ----------------------------
constexpr int NWAVE    = 4;                           // 128-thread blocks
constexpr int OFF_WHH   = 0;                          // 128 frags * 1KB
constexpr int OFF_WIH   = OFF_WHH   + 128 * 1024;     //  32 frags
constexpr int OFF_WH2P  = OFF_WIH   + 32 * 1024;      //   4 frags
constexpr int OFF_WSE   = OFF_WH2P  + 4 * 1024;       //   1 frag
constexpr int OFF_WCONF = OFF_WSE   + 1 * 1024;       //   4 frags
constexpr int OFF_BIASG = OFF_WCONF + 4 * 1024;       // 512 f32
constexpr int OFF_BH2P  = OFF_BIASG + 512 * 4;
constexpr int OFF_BSE   = OFF_BH2P  + 64;
constexpr int OFF_BCONF = OFF_BSE   + 64;
constexpr int OFF_HBUF  = OFF_BCONF + 64;             // 4 waves * 32x128 bf16 = 32KB
constexpr int OFF_RELB  = OFF_HBUF  + NWAVE * 4096 * 2;  // 4 * 32x32 bf16 = 8KB
constexpr int OFF_DECB  = OFF_RELB  + NWAVE * 1024 * 2;  // 8KB
constexpr int OFF_CONFB = OFF_DECB  + NWAVE * 1024 * 2;  // 4 * 32x8 f32 = 4KB
constexpr int LDS_BYTES = OFF_CONFB + NWAVE * 256 * 4;   // 228544 (< 320KB/WGP)

constexpr int HD = 128, TSTEPS = 30;

__global__ __launch_bounds__(128) void mm_lstm_kernel(
    const float* __restrict__ rel,     const float* __restrict__ h0,
    const float* __restrict__ c0,
    const float* __restrict__ W_se,    const float* __restrict__ b_se,
    const float* __restrict__ W_ih,    const float* __restrict__ W_hh,
    const float* __restrict__ b_ih,    const float* __restrict__ b_hh,
    const float* __restrict__ W_h2p,   const float* __restrict__ b_h2p,
    const float* __restrict__ W_conf,  const float* __restrict__ b_conf,
    float* __restrict__ out_traj,      float* __restrict__ out_conf)
{
    extern __shared__ __align__(16) char smem[];
    __bf16* whh   = (__bf16*)(smem + OFF_WHH);
    __bf16* wih   = (__bf16*)(smem + OFF_WIH);
    __bf16* wh2p  = (__bf16*)(smem + OFF_WH2P);
    __bf16* wse   = (__bf16*)(smem + OFF_WSE);
    __bf16* wconf = (__bf16*)(smem + OFF_WCONF);
    float* bias_g    = (float*)(smem + OFF_BIASG);
    float* bias_h2p  = (float*)(smem + OFF_BH2P);
    float* bias_se   = (float*)(smem + OFF_BSE);
    float* bias_conf = (float*)(smem + OFF_BCONF);

    const int tid  = threadIdx.x;
    const int wave = tid >> 5;
    const int lane = tid & 31;
    const int col  = lane & 15;        // N column (C layout) / M row (A layout)
    const int half = lane >> 4;
    const int Mrow = half * 8;         // C-layout row base for this lane
    const int bbase = blockIdx.x * 128 + wave * 32;   // 32 rows per wave

    __bf16* hb    = (__bf16*)(smem + OFF_HBUF) + wave * 4096;  // 32x128 bf16
    __bf16* relb  = (__bf16*)(smem + OFF_RELB) + wave * 1024;  // 32x32 bf16
    __bf16* decb  = (__bf16*)(smem + OFF_DECB) + wave * 1024;  // 32x32 bf16
    float*  confb = (float*)(smem + OFF_CONFB) + wave * 256;   // 32x8 f32

    // ---------------- one-time weight convert + swizzle into LDS ----------
    // Lane chunk of B-frag (kt,nt) = 16 consecutive k of row n=nt*16+lane%16,
    // k0 = kt*32 + (lane>=16 ? 16 : 0)  (contiguous k order; A frags use the same).
    for (int ch = tid; ch < 128 * 32; ch += 128) {             // W_hh (512x128)
        int frag = ch >> 5, l = ch & 31;
        int nt = frag >> 2, kt = frag & 3;
        int n  = nt * 16 + (l & 15);
        int k0 = kt * 32 + (l >> 4) * 16;
        __bf16* dst = whh + ch * 16;
        const float* src = W_hh + n * 128 + k0;
#pragma unroll
        for (int e = 0; e < 16; ++e) dst[e] = (__bf16)src[e];
    }
    for (int ch = tid; ch < 32 * 32; ch += 128) {              // W_ih (512x16), K pad->32
        int nt = ch >> 5, l = ch & 31;
        int n = nt * 16 + (l & 15);
        __bf16* dst = wih + ch * 16;
#pragma unroll
        for (int e = 0; e < 16; ++e)
            dst[e] = (l < 16) ? (__bf16)W_ih[n * 16 + e] : (__bf16)0.0f;
    }
    for (int ch = tid; ch < 4 * 32; ch += 128) {               // W_h2p (12x128), N pad->16
        int kt = ch >> 5, l = ch & 31;
        int n = l & 15, k0 = kt * 32 + (l >> 4) * 16;
        __bf16* dst = wh2p + ch * 16;
#pragma unroll
        for (int e = 0; e < 16; ++e)
            dst[e] = (n < 12) ? (__bf16)W_h2p[n * 128 + k0 + e] : (__bf16)0.0f;
    }
    for (int ch = tid; ch < 32; ch += 128) {                   // W_se (16x12), K pad->32
        int l = ch;
        int n = l & 15, k0 = (l >> 4) * 16;
        __bf16* dst = wse + ch * 16;
#pragma unroll
        for (int e = 0; e < 16; ++e)
            dst[e] = (k0 + e < 12) ? (__bf16)W_se[n * 12 + k0 + e] : (__bf16)0.0f;
    }
    for (int ch = tid; ch < 4 * 32; ch += 128) {               // W_conf (6x128), N pad->16
        int kt = ch >> 5, l = ch & 31;
        int n = l & 15, k0 = kt * 32 + (l >> 4) * 16;
        __bf16* dst = wconf + ch * 16;
#pragma unroll
        for (int e = 0; e < 16; ++e)
            dst[e] = (n < 6) ? (__bf16)W_conf[n * 128 + k0 + e] : (__bf16)0.0f;
    }
    for (int i = tid; i < 512; i += 128) bias_g[i] = b_ih[i] + b_hh[i];
    if (tid < 16) {
        bias_h2p[tid]  = (tid < 12) ? b_h2p[tid]  : 0.0f;
        bias_se[tid]   = b_se[tid];
        bias_conf[tid] = (tid < 6)  ? b_conf[tid] : 0.0f;
    }
    __syncthreads();

    // ---------------- per-wave state init ---------------------------------
    for (int idx = lane; idx < 4096; idx += 32)                // h -> LDS (bf16)
        hb[idx] = (__bf16)h0[(size_t)bbase * HD + idx];

    v8f cacc[2][8];                                            // c in VGPRs
#pragma unroll
    for (int s = 0; s < 2; ++s)
#pragma unroll
        for (int j = 0; j < 8; ++j)
#pragma unroll
            for (int v = 0; v < 8; ++v)
                cacc[s][j][v] =
                    c0[(size_t)(bbase + s * 16 + Mrow + v) * HD + j * 16 + col];

    // x0 = tile(last_obs_rel, MODES) -> relbuf rows (K=12); one row per lane
    {
        float x = rel[(size_t)(bbase + lane) * 2 + 0];
        float y = rel[(size_t)(bbase + lane) * 2 + 1];
        __bf16* rp = relb + lane * 32;
#pragma unroll
        for (int m = 0; m < 6; ++m) { rp[2 * m] = (__bf16)x; rp[2 * m + 1] = (__bf16)y; }
    }

    v16bf ax[2];        // dec_in A fragments (one per 16-row sub-tile)
    v16bf ah[2][4];     // h A fragments

    // dec_in = leaky(relbuf @ W_se^T + b_se) -> decbuf -> A fragments
    auto dec_step = [&]() {
        v16bf bw = ldB(wse + lane * 16);
        float bs = bias_se[col];
#pragma unroll
        for (int s = 0; s < 2; ++s) {
            const __bf16* rrow = relb + (s * 16 + col) * 32;
            v16bf ar = ldA(rrow + half * 16, rrow + half * 16 + 8);
            v8f dc;
#pragma unroll
            for (int v = 0; v < 8; ++v) dc[v] = bs;
            dc = wmma_bf16(ar, bw, dc);
#pragma unroll
            for (int v = 0; v < 8; ++v)
                decb[(s * 16 + Mrow + v) * 32 + col] = (__bf16)leaky_(dc[v]);
        }
#pragma unroll
        for (int s = 0; s < 2; ++s) {
            const __bf16* drow = decb + (s * 16 + col) * 32;
            ax[s] = ldA(drow + half * 16, drow + half * 16 + 8);
        }
    };

    auto load_ah = [&]() {
#pragma unroll
        for (int s = 0; s < 2; ++s) {
            const __bf16* hrow = hb + (s * 16 + col) * HD;
#pragma unroll
            for (int kt = 0; kt < 4; ++kt)
                ah[s][kt] = ldA(hrow + kt * 32 + half * 16,
                                hrow + kt * 32 + half * 16 + 8);
        }
    };

    // one gate region (16 cols) for both sub-tiles; B frag shared across s
    auto gate = [&](int nt, v8f (&out)[2]) {
        float bg = bias_g[nt * 16 + col];
        v8f a0, a1;
#pragma unroll
        for (int v = 0; v < 8; ++v) { a0[v] = bg; a1[v] = bg; }
        v16bf b = ldB(wih + (nt * 32 + lane) * 16);
        a0 = wmma_bf16(ax[0], b, a0);
        a1 = wmma_bf16(ax[1], b, a1);
#pragma unroll
        for (int kt = 0; kt < 4; ++kt) {
            v16bf bb = ldB(whh + ((nt * 4 + kt) * 32 + lane) * 16);
            a0 = wmma_bf16(ah[0][kt], bb, a0);
            a1 = wmma_bf16(ah[1][kt], bb, a1);
        }
        out[0] = a0; out[1] = a1;
    };

    dec_step();
    load_ah();

    // ---------------- recurrent loop (kept rolled: huge body) -------------
#pragma unroll 1
    for (int t = 0; t < TSTEPS; ++t) {
#pragma unroll
        for (int j = 0; j < 8; ++j) {                          // H tile 16 cols
            v8f g0[2], g1[2], p[2];
            gate(0 * 8 + j, g0);                               // i
            gate(2 * 8 + j, g1);                               // g
#pragma unroll
            for (int s = 0; s < 2; ++s)
#pragma unroll
                for (int v = 0; v < 8; ++v)
                    p[s][v] = sigmoid_t(g0[s][v]) * tanh_t(g1[s][v]);
            gate(1 * 8 + j, g0);                               // f
#pragma unroll
            for (int s = 0; s < 2; ++s)
#pragma unroll
                for (int v = 0; v < 8; ++v) {
                    float cn = sigmoid_t(g0[s][v]) * cacc[s][j][v] + p[s][v];
                    cacc[s][j][v] = cn;
                    p[s][v] = tanh_t(cn);
                }
            gate(3 * 8 + j, g1);                               // o
#pragma unroll
            for (int s = 0; s < 2; ++s)
#pragma unroll
                for (int v = 0; v < 8; ++v)
                    hb[(s * 16 + Mrow + v) * HD + j * 16 + col] =
                        (__bf16)(sigmoid_t(g1[s][v]) * p[s][v]);
        }
        load_ah();                                             // updated h -> A frags

        // rel_pos = h @ W_h2p^T + b  (N=12 in a 16-wide tile); B frag shared
        v8f rp[2];
        float bh = bias_h2p[col];
#pragma unroll
        for (int s = 0; s < 2; ++s)
#pragma unroll
            for (int v = 0; v < 8; ++v) rp[s][v] = bh;
#pragma unroll
        for (int kt = 0; kt < 4; ++kt) {
            v16bf bb = ldB(wh2p + (kt * 32 + lane) * 16);
            rp[0] = wmma_bf16(ah[0][kt], bb, rp[0]);
            rp[1] = wmma_bf16(ah[1][kt], bb, rp[1]);
        }
        // scatter into pred_traj (B,6,30,2); coalesces in L2 before writeback
        if (col < 12) {
#pragma unroll
            for (int s = 0; s < 2; ++s)
#pragma unroll
                for (int v = 0; v < 8; ++v)
                    out_traj[(size_t)(bbase + s * 16 + Mrow + v) * 360 +
                             (col >> 1) * 60 + t * 2 + (col & 1)] = rp[s][v];
        }
        // feed rel_pos back through the spatial embedding
#pragma unroll
        for (int s = 0; s < 2; ++s)
#pragma unroll
            for (int v = 0; v < 8; ++v)
                relb[(s * 16 + Mrow + v) * 32 + col] = (__bf16)rp[s][v];
        dec_step();
    }

    // ---------------- confidences: softmax(h @ W_conf^T + b) --------------
    v8f cf[2];
    float bc = bias_conf[col];
#pragma unroll
    for (int s = 0; s < 2; ++s)
#pragma unroll
        for (int v = 0; v < 8; ++v) cf[s][v] = bc;
#pragma unroll
    for (int kt = 0; kt < 4; ++kt) {
        v16bf bb = ldB(wconf + (kt * 32 + lane) * 16);
        cf[0] = wmma_bf16(ah[0][kt], bb, cf[0]);
        cf[1] = wmma_bf16(ah[1][kt], bb, cf[1]);
    }
    if (col < 6) {
#pragma unroll
        for (int s = 0; s < 2; ++s)
#pragma unroll
            for (int v = 0; v < 8; ++v)
                confb[(s * 16 + Mrow + v) * 8 + col] = cf[s][v];
    }
    {   // one row per lane (32 rows)
        float vv[6], m = -1e30f;
#pragma unroll
        for (int i = 0; i < 6; ++i) { vv[i] = confb[lane * 8 + i]; m = fmaxf(m, vv[i]); }
        float ssum = 0.0f;
#pragma unroll
        for (int i = 0; i < 6; ++i) { vv[i] = __expf(vv[i] - m); ssum += vv[i]; }
        float inv = 1.0f / ssum;
#pragma unroll
        for (int i = 0; i < 6; ++i)
            out_conf[(size_t)(bbase + lane) * 6 + i] = vv[i] * inv;
    }
}

extern "C" void kernel_launch(void* const* d_in, const int* in_sizes, int n_in,
                              void* d_out, int out_size, void* d_ws, size_t ws_size,
                              hipStream_t stream) {
    (void)n_in; (void)d_ws; (void)ws_size; (void)out_size;
    const float* rel    = (const float*)d_in[1];   // last_obs_rel; d_in[0] unused
    const float* h0     = (const float*)d_in[2];
    const float* c0     = (const float*)d_in[3];
    const float* W_se   = (const float*)d_in[4];
    const float* b_se   = (const float*)d_in[5];
    const float* W_ih   = (const float*)d_in[6];
    const float* W_hh   = (const float*)d_in[7];
    const float* b_ih   = (const float*)d_in[8];
    const float* b_hh   = (const float*)d_in[9];
    const float* W_h2p  = (const float*)d_in[10];
    const float* b_h2p  = (const float*)d_in[11];
    const float* W_conf = (const float*)d_in[12];
    const float* b_conf = (const float*)d_in[13];

    const int B = in_sizes[1] / 2;                 // 65536
    float* out_traj = (float*)d_out;
    float* out_conf = out_traj + (size_t)B * 360;

    (void)hipFuncSetAttribute((const void*)mm_lstm_kernel,
                              hipFuncAttributeMaxDynamicSharedMemorySize, LDS_BYTES);

    dim3 grid(B / 128), block(128);
    mm_lstm_kernel<<<grid, block, LDS_BYTES, stream>>>(
        rel, h0, c0, W_se, b_se, W_ih, W_hh, b_ih, b_hh,
        W_h2p, b_h2p, W_conf, b_conf, out_traj, out_conf);
}
